// SceneGraphEncoderV2_49289044689186
// MI455X (gfx1250) — matrix operand order repaired and checked by reading, not compiled, not run on hardware
//
#include <hip/hip_runtime.h>

// ---------------------------------------------------------------------------
// SceneGraphEncoderV2 for MI455X (gfx1250, wave32, WMMA).
// All GEMMs (combiner, Q/K/V/O, FFN, Q.K^T, attn.V) use
// v_wmma_f32_16x16x32_f16 (f16 inputs, f32 accumulate).
// Relation gates are pre-packed into a u16 bitmask tensor (coalesced single
// pass over the 84MB relations tensor; 4MB result stays L2-resident).
// ---------------------------------------------------------------------------

#define B_   8
#define N_   512
#define D_   192
#define R_   10
#define H_   12
#define DH_  16
#define ROWS (B_ * N_)           // 4096
#define FF_  (2 * D_)            // 384

typedef _Float16 h8  __attribute__((ext_vector_type(8)));
typedef _Float16 v16h __attribute__((ext_vector_type(16)));
typedef float    v8f __attribute__((ext_vector_type(8)));

__device__ __forceinline__ v16h cat16(h8 lo, h8 hi) {
  return __builtin_shufflevector(lo, hi, 0,1,2,3,4,5,6,7,8,9,10,11,12,13,14,15);
}
__device__ __forceinline__ float gelu_exact(float x) {
  return 0.5f * x * (1.0f + erff(x * 0.70710678118654752440f));
}

// ---------------------------------------------------------------------------
// Weight transpose + f32 -> f16 convert: WT[n*K+k] = (f16) W[k*N+n]
// ---------------------------------------------------------------------------
__global__ void wtrans_kernel(const float* __restrict__ W, _Float16* __restrict__ WT,
                              int K, int N) {
  int idx = blockIdx.x * blockDim.x + threadIdx.x;
  if (idx >= K * N) return;
  int n = idx / K, k = idx % K;
  WT[idx] = (_Float16)W[(long)k * N + n];
}

// ---------------------------------------------------------------------------
// Pack relation gates + key mask into u16 bitmask per (b, nq, nk).
// bits 0..9 : relation[b,nq,nk,h] >= 1e-4  (keep typed head h)
// bit  15   : node_mask[b,nk] > 0          (key valid)
// Fully coalesced streaming read of the 84MB relations tensor.
// ---------------------------------------------------------------------------
__global__ void relmask_kernel(const float* __restrict__ rel,
                               const float* __restrict__ node_mask,
                               unsigned short* __restrict__ relmask) {
  long idx = (long)blockIdx.x * blockDim.x + threadIdx.x;   // (b*512+nq)*512+nk
  if (idx >= (long)B_ * N_ * N_) return;
  const float* gp = rel + idx * R_;
  unsigned m = 0;
  #pragma unroll
  for (int j = 0; j < R_; ++j) m |= (gp[j] >= 1e-4f) ? (1u << j) : 0u;
  int nk = (int)(idx & (N_ - 1));
  int b  = (int)(idx >> 18);               // N_*N_ == 2^18
  if (node_mask[b * N_ + nk] > 0.f) m |= 0x8000u;
  relmask[idx] = (unsigned short)m;
}

// ---------------------------------------------------------------------------
// Front end: embeddings + bbox proj (8->48, GELU, LN) -> combined (ROWS x 192)
// ---------------------------------------------------------------------------
__global__ void build_combined_kernel(const int* __restrict__ region_ids,
                                      const int* __restrict__ entity_ids,
                                      const int* __restrict__ pos_ids,
                                      const float* __restrict__ bbox_feats,
                                      const float* __restrict__ region_emb,
                                      const float* __restrict__ entity_emb,
                                      const float* __restrict__ pos_emb,
                                      const float* __restrict__ bbox_w,
                                      const float* __restrict__ bbox_b,
                                      const float* __restrict__ ln_g,
                                      const float* __restrict__ ln_b,
                                      float* __restrict__ combined,
                                      _Float16* __restrict__ combinedH) {
  int r = blockIdx.x * blockDim.x + threadIdx.x;
  if (r >= ROWS) return;
  float* out = combined + (long)r * D_;
  int rid = region_ids[r], eid = entity_ids[r], pid = pos_ids[r];
  for (int j = 0; j < 64; ++j) out[j]       = region_emb[(long)rid * 64 + j];
  for (int j = 0; j < 64; ++j) out[64 + j]  = entity_emb[(long)eid * 64 + j];
  for (int j = 0; j < 16; ++j) out[128 + j] = pos_emb[(long)pid * 16 + j];
  const float* x = bbox_feats + (long)r * 8;
  float e[48]; float s = 0.f, s2 = 0.f;
  for (int j = 0; j < 48; ++j) {
    float acc = bbox_b[j];
    #pragma unroll
    for (int i = 0; i < 8; ++i) acc += x[i] * bbox_w[i * 48 + j];
    acc = gelu_exact(acc);
    e[j] = acc; s += acc; s2 += acc * acc;
  }
  float mean = s * (1.f / 48.f);
  float var  = s2 * (1.f / 48.f) - mean * mean;
  float inv  = rsqrtf(var + 1e-5f);
  for (int j = 0; j < 48; ++j) out[144 + j] = (e[j] - mean) * inv * ln_g[j] + ln_b[j];
  for (int j = 0; j < D_; ++j) combinedH[(long)r * D_ + j] = (_Float16)out[j];
}

// ---------------------------------------------------------------------------
// WMMA GEMM: out = A(M,K f16) @ W + bias,  W given transposed as WT(N,K f16).
// One 16x16 C tile per wave, 8 waves / block.  EPI==1 -> exact GELU.
// ---------------------------------------------------------------------------
template <int EPI>
__global__ __launch_bounds__(256)
void gemm_bias_kernel(const _Float16* __restrict__ A, const _Float16* __restrict__ WT,
                      const float* __restrict__ bias,
                      float* __restrict__ outF, _Float16* __restrict__ outH,
                      int M, int N, int K) {
  int wave = threadIdx.x >> 5;
  int lane = threadIdx.x & 31;
  int tilesN = N >> 4;
  int tilesTot = (M >> 4) * tilesN;
  int tile = blockIdx.x * 8 + wave;          // wave-uniform
  if (tile >= tilesTot) return;              // uniform exit: EXEC all-1 for WMMA
  int tm = tile / tilesN, tn = tile % tilesN;
  int g = lane >> 4, r = lane & 15;
  const _Float16* Arow = A  + (long)(tm * 16 + r) * K;
  const _Float16* Brow = WT + (long)(tn * 16 + r) * K;
  v8f c = {};
  for (int k0 = 0; k0 < K; k0 += 32) {
    h8 a_lo = *(const h8*)(Arow + k0 + 8 * g);
    h8 a_hi = *(const h8*)(Arow + k0 + 16 + 8 * g);
    h8 b_lo = *(const h8*)(Brow + k0 + 8 * g);
    h8 b_hi = *(const h8*)(Brow + k0 + 16 + 8 * g);
    c = __builtin_amdgcn_wmma_f32_16x16x32_f16(false, cat16(a_lo, a_hi),
                                               false, cat16(b_lo, b_hi),
                                               (short)0, c, false, false);
  }
  float bn = bias ? bias[tn * 16 + r] : 0.f;
  #pragma unroll
  for (int i = 0; i < 8; ++i) {
    float x = c[i] + bn;
    if (EPI == 1) x = gelu_exact(x);
    long m = tm * 16 + i + 8 * g;
    long n = tn * 16 + r;
    if (outF) outF[m * N + n] = x;
    if (outH) outH[m * N + n] = (_Float16)x;
  }
}

// ---------------------------------------------------------------------------
// Row LayerNorm (wave per row): y = LN(x [+ res]) * g + b   [optional GELU]
// ---------------------------------------------------------------------------
__global__ __launch_bounds__(256)
void addln_kernel(const float* __restrict__ x, const float* __restrict__ res,
                  const float* __restrict__ g, const float* __restrict__ b,
                  float* __restrict__ outF, _Float16* __restrict__ outH,
                  int rows, int cols, int gelu_after) {
  int wave = threadIdx.x >> 5, lane = threadIdx.x & 31;
  int row = blockIdx.x * 8 + wave;
  if (row >= rows) return;
  const float* xr = x + (long)row * cols;
  const float* rr = res ? res + (long)row * cols : nullptr;
  float s = 0.f, s2 = 0.f;
  for (int c = lane; c < cols; c += 32) {
    float v = xr[c] + (rr ? rr[c] : 0.f);
    s += v; s2 += v * v;
  }
  for (int m = 16; m >= 1; m >>= 1) { s += __shfl_xor(s, m, 32); s2 += __shfl_xor(s2, m, 32); }
  float mean = s / cols;
  float var  = s2 / cols - mean * mean;
  float inv  = rsqrtf(var + 1e-5f);
  for (int c = lane; c < cols; c += 32) {
    float v = xr[c] + (rr ? rr[c] : 0.f);
    float y = (v - mean) * inv * g[c] + b[c];
    if (gelu_after) y = gelu_exact(y);
    if (outF) outF[(long)row * cols + c] = y;
    if (outH) outH[(long)row * cols + c] = (_Float16)y;
  }
}

// ---------------------------------------------------------------------------
// Pack q,k (B,N,H,DH) f32 -> (B,H,N,DH) f16   (coalesced reads)
// ---------------------------------------------------------------------------
__global__ void pack_qk_kernel(const float* __restrict__ q, const float* __restrict__ k,
                               _Float16* __restrict__ qH, _Float16* __restrict__ kH) {
  int idx = blockIdx.x * blockDim.x + threadIdx.x;
  if (idx >= ROWS * D_) return;
  int dh = idx & 15;
  int h  = (idx >> 4) % H_;
  int n  = (idx / D_) % N_;
  int b  = idx / (D_ * N_);
  long p = (((long)b * H_ + h) * N_ + n) * DH_ + dh;
  qH[p] = (_Float16)q[idx];
  kH[p] = (_Float16)k[idx];
}

// Pack v (B,N,H,DH) f32 -> vT (B,H,DH,N) f16  (coalesced writes: n fastest)
__global__ void pack_vT_kernel(const float* __restrict__ v, _Float16* __restrict__ vT) {
  int idx = blockIdx.x * blockDim.x + threadIdx.x;    // (b,h,dh,n), n fastest
  if (idx >= ROWS * D_) return;
  int n  = idx & (N_ - 1);
  int dh = (idx >> 9) & 15;
  int h  = (idx / (N_ * DH_)) % H_;
  int b  = idx / (N_ * DH_ * H_);
  vT[idx] = (_Float16)v[((long)b * N_ + n) * D_ + h * DH_ + dh];
}

// ---------------------------------------------------------------------------
// Attention: one block per (b, h, 16-query tile).  8 waves / 256 threads.
//   phase 1: scores = (Q K^T)/4 via WMMA (DH=16 zero-padded to K=32),
//            u16 bitmask gating (typed-head relation bit + key-valid bit),
//            branch-free selects -> LDS (16 x 512)
//   phase 2: masked softmax (fully-masked rows -> 0), probs -> f16 LDS
//   phase 3: attn @ V via WMMA (A from LDS, B from vT), 8-way partial reduce
// ---------------------------------------------------------------------------
__global__ __launch_bounds__(256)
void attention_kernel(const _Float16* __restrict__ qH, const _Float16* __restrict__ kH,
                      const _Float16* __restrict__ vT,
                      const unsigned short* __restrict__ relmask,
                      _Float16* __restrict__ attnH /* (B,N,D) f16 */) {
  __shared__ float    sc[16][N_];          // 32 KB
  __shared__ _Float16 pr[16][N_];          // 16 KB
  __shared__ float    part[8][16][16];     //  8 KB
  int wave = threadIdx.x >> 5, lane = threadIdx.x & 31;
  int qt = blockIdx.x & 31;
  int bh = blockIdx.x >> 5;                // b*H + h
  int h = bh % H_, b = bh / H_;
  int g = lane >> 4, r = lane & 15;
  const unsigned need = (h < R_) ? (1u << h) : 0u;   // typed head bit, 0 = shared

  // A fragment: Q tile (16 x 32, upper K half zero)
  const _Float16* qrow = qH + ((long)bh * N_ + qt * 16 + r) * DH_;
  h8 zero8 = {};
  v16h afrag = cat16(*(const h8*)(qrow + 8 * g), zero8);

  const float NEG = -__builtin_inff();
  for (int kt = wave; kt < N_ / 16; kt += 8) {
    h8 b_lo, b_hi = zero8;
    {                                      // only K(dh)=0..15 is real (s<8 slots)
      const _Float16* krow = kH + ((long)bh * N_ + kt * 16 + r) * DH_;
      b_lo = *(const h8*)(krow + 8 * g);
    }
    // gate bits for the 8 C elements of this lane: rows nq = qt*16 + i + 8g
    const unsigned short* mrow =
        relmask + ((long)b * N_ + qt * 16 + 8 * g) * N_ + kt * 16 + r;
    unsigned mm[8];
    #pragma unroll
    for (int i = 0; i < 8; ++i) mm[i] = mrow[(long)i * N_];   // independent, clause-able
    v8f c = {};
    c = __builtin_amdgcn_wmma_f32_16x16x32_f16(false, afrag, false, cat16(b_lo, b_hi),
                                               (short)0, c, false, false);
    int nk = kt * 16 + r;
    #pragma unroll
    for (int i = 0; i < 8; ++i) {
      float s = c[i] * 0.25f;              // 1/sqrt(16)
      bool ok = (mm[i] & 0x8000u) && ((mm[i] & need) == need);
      sc[i + 8 * g][nk] = ok ? s : NEG;    // branch-free select
    }
  }
  __syncthreads();

  // softmax: each wave handles rows {wave, wave+8}
  for (int row = wave; row < 16; row += 8) {
    float mx = NEG;
    for (int c = lane; c < N_; c += 32) mx = fmaxf(mx, sc[row][c]);
    for (int m = 16; m >= 1; m >>= 1) mx = fmaxf(mx, __shfl_xor(mx, m, 32));
    float sum = 0.f;
    for (int c = lane; c < N_; c += 32) {
      float e = (mx == NEG) ? 0.f : __expf(sc[row][c] - mx);
      sc[row][c] = e; sum += e;
    }
    for (int m = 16; m >= 1; m >>= 1) sum += __shfl_xor(sum, m, 32);
    float invs = (sum > 0.f) ? 1.f / sum : 0.f;      // nan_to_num behaviour
    for (int c = lane; c < N_; c += 32) pr[row][c] = (_Float16)(sc[row][c] * invs);
  }
  __syncthreads();

  // attn @ V : K(keys)=512 = 16 steps of 32; each wave accumulates 2 steps
  v8f c2 = {};
  #pragma unroll
  for (int j = 0; j < 2; ++j) {
    int k0 = (wave * 2 + j) * 32;
    h8 a0 = *(const h8*)(&pr[r][k0 + 8 * g]);
    h8 a1 = *(const h8*)(&pr[r][k0 + 16 + 8 * g]);
    const _Float16* vrow = vT + ((long)bh * DH_ + r) * N_ + k0;
    h8 b0 = *(const h8*)(vrow + 8 * g);
    h8 b1 = *(const h8*)(vrow + 16 + 8 * g);
    c2 = __builtin_amdgcn_wmma_f32_16x16x32_f16(false, cat16(a0, a1),
                                                false, cat16(b0, b1),
                                                (short)0, c2, false, false);
  }
  #pragma unroll
  for (int i = 0; i < 8; ++i) part[wave][i + 8 * g][r] = c2[i];
  __syncthreads();

  int mrow2 = threadIdx.x >> 4, nc = threadIdx.x & 15;   // 256 = 16x16
  float acc = 0.f;
  #pragma unroll
  for (int w = 0; w < 8; ++w) acc += part[w][mrow2][nc];
  attnH[((long)b * N_ + qt * 16 + mrow2) * D_ + h * DH_ + nc] = (_Float16)acc;
}

__global__ void copy_f32_kernel(const float* __restrict__ a, float* __restrict__ b, int n) {
  int i = blockIdx.x * blockDim.x + threadIdx.x;
  if (i < n) b[i] = a[i];
}

// ---------------------------------------------------------------------------
// Host-side orchestration
// ---------------------------------------------------------------------------
// Input index map.  Top-level setup_inputs() insertion order; the nested
// "params" pytree expanded as leaves with sorted dict keys (JAX pytree order),
// layers list in order.
enum {
  IN_REGION_IDS = 0, IN_ENTITY_IDS, IN_POS_IDS, IN_BBOX, IN_REL, IN_MASK,
  P_BBOX_B, P_BBOX_LN_B, P_BBOX_LN_G, P_BBOX_W,
  P_COMB_B, P_COMB_LN_B, P_COMB_LN_G, P_COMB_W,
  P_ENTITY_EMB,
  P_L0 = 15,                 // 16 leaves per layer, 2 layers
  P_POS_EMB = 15 + 32, P_REGION_EMB
};
// per-layer leaf offsets (sorted keys)
enum { L_BK = 0, L_BO, L_BQ, L_BV, L_FFN_B1, L_FFN_B2, L_FFN_W1, L_FFN_W2,
       L_LN1_B, L_LN1_G, L_LN2_B, L_LN2_G, L_WK, L_WO, L_WQ, L_WV };

static inline void launch_gemm(int epi, const _Float16* A, const _Float16* WT,
                               const float* bias, float* outF, _Float16* outH,
                               int M, int N, int K, hipStream_t s) {
  int tiles = (M / 16) * (N / 16);
  int blocks = (tiles + 7) / 8;
  if (epi) gemm_bias_kernel<1><<<blocks, 256, 0, s>>>(A, WT, bias, outF, outH, M, N, K);
  else     gemm_bias_kernel<0><<<blocks, 256, 0, s>>>(A, WT, bias, outF, outH, M, N, K);
}

extern "C" void kernel_launch(void* const* d_in, const int* in_sizes, int n_in,
                              void* d_out, int out_size, void* d_ws, size_t ws_size,
                              hipStream_t stream) {
  (void)in_sizes; (void)n_in; (void)out_size; (void)ws_size;
  const int*   region_ids = (const int*)d_in[IN_REGION_IDS];
  const int*   entity_ids = (const int*)d_in[IN_ENTITY_IDS];
  const int*   pos_ids    = (const int*)d_in[IN_POS_IDS];
  const float* bbox_feats = (const float*)d_in[IN_BBOX];
  const float* relations  = (const float*)d_in[IN_REL];
  const float* node_mask  = (const float*)d_in[IN_MASK];

  // workspace bump allocator (256B aligned)
  char* wsp = (char*)d_ws;
  auto alloc = [&](size_t bytes) -> void* {
    void* p = wsp; wsp += (bytes + 255) & ~(size_t)255; return p;
  };
  float*    combined  = (float*)   alloc(sizeof(float) * ROWS * D_);
  _Float16* combinedH = (_Float16*)alloc(sizeof(_Float16) * ROWS * D_);
  float*    tmp       = (float*)   alloc(sizeof(float) * ROWS * FF_);
  float*    nodesF0   = (float*)   alloc(sizeof(float) * ROWS * D_);
  float*    nodesF1   = (float*)   alloc(sizeof(float) * ROWS * D_);
  _Float16* nodesH0   = (_Float16*)alloc(sizeof(_Float16) * ROWS * D_);
  _Float16* nodesH1   = (_Float16*)alloc(sizeof(_Float16) * ROWS * D_);
  float*    hdd       = (float*)   alloc(sizeof(float) * ROWS * D_);
  _Float16* hddH      = (_Float16*)alloc(sizeof(_Float16) * ROWS * D_);
  float*    qF        = (float*)   alloc(sizeof(float) * ROWS * D_);
  float*    kF        = (float*)   alloc(sizeof(float) * ROWS * D_);
  float*    vF        = (float*)   alloc(sizeof(float) * ROWS * D_);
  _Float16* qH        = (_Float16*)alloc(sizeof(_Float16) * ROWS * D_);
  _Float16* kH        = (_Float16*)alloc(sizeof(_Float16) * ROWS * D_);
  _Float16* vT        = (_Float16*)alloc(sizeof(_Float16) * ROWS * D_);
  _Float16* attnH     = (_Float16*)alloc(sizeof(_Float16) * ROWS * D_);
  _Float16* ffH       = (_Float16*)alloc(sizeof(_Float16) * ROWS * FF_);
  unsigned short* relmask = (unsigned short*)alloc(sizeof(unsigned short) * (size_t)B_ * N_ * N_);
  _Float16* comb_wT   = (_Float16*)alloc(sizeof(_Float16) * D_ * D_);
  _Float16* lwT[2][6];  // wq,wk,wv,wo,w1,w2 transposed per layer
  for (int l = 0; l < 2; ++l) {
    for (int j = 0; j < 4; ++j) lwT[l][j] = (_Float16*)alloc(sizeof(_Float16) * D_ * D_);
    lwT[l][4] = (_Float16*)alloc(sizeof(_Float16) * D_ * FF_);
    lwT[l][5] = (_Float16*)alloc(sizeof(_Float16) * FF_ * D_);
  }

  // ---- relation gate bitmask (once; reused by both layers) ----
  {
    long e = (long)B_ * N_ * N_;
    relmask_kernel<<<(int)((e + 255) / 256), 256, 0, stream>>>(relations, node_mask, relmask);
  }

  // ---- weight transposes / f16 convert ----
  auto trans = [&](const float* W, _Float16* WT, int K, int N) {
    int e = K * N;
    wtrans_kernel<<<(e + 255) / 256, 256, 0, stream>>>(W, WT, K, N);
  };
  trans((const float*)d_in[P_COMB_W], comb_wT, D_, D_);
  for (int l = 0; l < 2; ++l) {
    int base = P_L0 + l * 16;
    trans((const float*)d_in[base + L_WQ],     lwT[l][0], D_, D_);
    trans((const float*)d_in[base + L_WK],     lwT[l][1], D_, D_);
    trans((const float*)d_in[base + L_WV],     lwT[l][2], D_, D_);
    trans((const float*)d_in[base + L_WO],     lwT[l][3], D_, D_);
    trans((const float*)d_in[base + L_FFN_W1], lwT[l][4], D_, FF_);
    trans((const float*)d_in[base + L_FFN_W2], lwT[l][5], FF_, D_);
  }

  // ---- front end ----
  build_combined_kernel<<<ROWS / 256, 256, 0, stream>>>(
      region_ids, entity_ids, pos_ids, bbox_feats,
      (const float*)d_in[P_REGION_EMB], (const float*)d_in[P_ENTITY_EMB],
      (const float*)d_in[P_POS_EMB],
      (const float*)d_in[P_BBOX_W], (const float*)d_in[P_BBOX_B],
      (const float*)d_in[P_BBOX_LN_G], (const float*)d_in[P_BBOX_LN_B],
      combined, combinedH);
  launch_gemm(0, combinedH, comb_wT, (const float*)d_in[P_COMB_B],
              tmp, nullptr, ROWS, D_, D_, stream);
  addln_kernel<<<ROWS / 8, 256, 0, stream>>>(tmp, nullptr,
      (const float*)d_in[P_COMB_LN_G], (const float*)d_in[P_COMB_LN_B],
      nodesF0, nodesH0, ROWS, D_, /*gelu=*/1);

  float*    curF = nodesF0; _Float16* curH = nodesH0;
  float*    nxtF = nodesF1; _Float16* nxtH = nodesH1;

  // ---- GAT layers ----
  for (int l = 0; l < 2; ++l) {
    int base = P_L0 + l * 16;
    launch_gemm(0, curH, lwT[l][0], (const float*)d_in[base + L_BQ], qF, nullptr, ROWS, D_, D_, stream);
    launch_gemm(0, curH, lwT[l][1], (const float*)d_in[base + L_BK], kF, nullptr, ROWS, D_, D_, stream);
    launch_gemm(0, curH, lwT[l][2], (const float*)d_in[base + L_BV], vF, nullptr, ROWS, D_, D_, stream);
    pack_qk_kernel<<<(ROWS * D_) / 256, 256, 0, stream>>>(qF, kF, qH, kH);
    pack_vT_kernel<<<(ROWS * D_) / 256, 256, 0, stream>>>(vF, vT);
    attention_kernel<<<B_ * H_ * (N_ / 16), 256, 0, stream>>>(qH, kH, vT, relmask, attnH);
    launch_gemm(0, attnH, lwT[l][3], (const float*)d_in[base + L_BO], tmp, nullptr, ROWS, D_, D_, stream);
    addln_kernel<<<ROWS / 8, 256, 0, stream>>>(tmp, curF,
        (const float*)d_in[base + L_LN1_G], (const float*)d_in[base + L_LN1_B],
        hdd, hddH, ROWS, D_, 0);
    launch_gemm(1, hddH, lwT[l][4], (const float*)d_in[base + L_FFN_B1],
                nullptr, ffH, ROWS, FF_, D_, stream);            // GELU epilogue
    launch_gemm(0, ffH, lwT[l][5], (const float*)d_in[base + L_FFN_B2],
                tmp, nullptr, ROWS, D_, FF_, stream);
    float* outF = (l == 1) ? (float*)d_out : nxtF;
    addln_kernel<<<ROWS / 8, 256, 0, stream>>>(tmp, hdd,
        (const float*)d_in[base + L_LN2_G], (const float*)d_in[base + L_LN2_B],
        outF, nxtH, ROWS, D_, 0);
    curF = outF; curH = nxtH;
    nxtF = (l == 0) ? nodesF0 : nxtF;   // ping-pong (layer count == 2)
    nxtH = (l == 0) ? nodesH0 : nxtH;
  }

  // second output: node_mask appended after nodes
  copy_f32_kernel<<<(ROWS + 255) / 256, 256, 0, stream>>>(
      node_mask, (float*)d_out + (long)ROWS * D_, ROWS);
}
